// GCN_55920474193966
// MI455X (gfx1250) — matrix operand order repaired and checked by reading, not compiled, not run on hardware
//
#include <hip/hip_runtime.h>

#define NN 50000
#define EE 800000
#define NF 128
#define NH 128
#define NC 40
#define NCPAD 48

typedef __attribute__((ext_vector_type(16))) __bf16 v16bf;
typedef __attribute__((ext_vector_type(8)))  __bf16 v8bf;
typedef __attribute__((ext_vector_type(4)))  __bf16 v4bf;
typedef __attribute__((ext_vector_type(8)))  float  v8f;
typedef __attribute__((ext_vector_type(4)))  float  v4f;

static inline int cdiv(long long a, long long b) { return (int)((a + b - 1) / b); }
static inline size_t alignup(size_t x) { return (x + 255) & ~(size_t)255; }

// ---------------- utility kernels ----------------

__global__ void fill_f32(float* __restrict__ p, float v, int n) {
    int i = blockIdx.x * blockDim.x + threadIdx.x;
    if (i < n) p[i] = v;
}

__global__ void deg_kernel(const int* __restrict__ ei, float* __restrict__ deg) {
    int e = blockIdx.x * blockDim.x + threadIdx.x;
    if (e >= EE) return;
    atomicAdd(&deg[ei[EE + e]], 1.0f);
}

__global__ void rsqrt_kernel(float* __restrict__ d, int n) {
    int i = blockIdx.x * blockDim.x + threadIdx.x;
    if (i >= n) return;
    float v = d[i];
    d[i] = (v > 0.0f) ? rsqrtf(v) : 0.0f;
}

__global__ void norm_kernel(const int* __restrict__ ei, const float* __restrict__ dinv,
                            float* __restrict__ nrm) {
    int e = blockIdx.x * blockDim.x + threadIdx.x;
    if (e >= EE) return;
    nrm[e] = dinv[ei[e]] * dinv[ei[EE + e]];
}

__global__ void cvt_bf16(const float* __restrict__ src, __bf16* __restrict__ dst, int n) {
    int i = blockIdx.x * blockDim.x + threadIdx.x;
    if (i < n) dst[i] = (__bf16)src[i];
}

// W is [128 x cols] row-major (k-major). Produce Wt [colsPad x 128] (n-major) bf16,
// zero-padded, so WMMA B fragments load as contiguous runs along K.
__global__ void cvt_w(const float* __restrict__ W, __bf16* __restrict__ Wt,
                      int cols, int colsPad) {
    int idx = blockIdx.x * blockDim.x + threadIdx.x;
    if (idx >= colsPad * 128) return;
    int n = idx / 128, k = idx - n * 128;
    float v = (n < cols) ? W[k * cols + n] : 0.0f;
    Wt[n * 128 + k] = (__bf16)v;
}

// ---------------- WMMA GEMM: C[N x colsPad] = A[N x 128] (bf16) @ Wt^T ----------------
// One wave per 16-row strip of A. All four K=32 A-fragments stay resident in VGPRs;
// the wave loops over N-tiles streaming B (tiny, L2/L0-resident weight matrix).
// 32 (or 12) WMMAs per wave, A global traffic amortized across all N-tiles.

__global__ void gemm_wmma(const __bf16* __restrict__ A, const __bf16* __restrict__ Wt,
                          float* __restrict__ C, int mTiles, int nTiles, int colsPad) {
    int wid = (blockIdx.x * blockDim.x + threadIdx.x) >> 5;  // wave32, wave-uniform
    if (wid >= mTiles) return;
    int lane = threadIdx.x & 31;
    int half = lane >> 4;  // 0: lanes 0-15, 1: lanes 16-31

    // A fragment (16x32 bf16): lane holds row M=lane%16; K runs {base..base+7} and
    // {base+16..base+23}, base = half*8 -> two contiguous 16B loads per K-block.
    int arow = wid * 16 + (lane & 15);
    const __bf16* ap = A + (size_t)arow * 128 + half * 8;
    v16bf a[4];
#pragma unroll
    for (int kk = 0; kk < 4; ++kk) {
        v8bf lo = *(const v8bf*)(ap + kk * 32);
        v8bf hi = *(const v8bf*)(ap + kk * 32 + 16);
#pragma unroll
        for (int i = 0; i < 8; ++i) { a[kk][i] = lo[i]; a[kk][i + 8] = hi[i]; }
    }

    // B fragment (32x16 bf16): lane holds col N=lane%16, K = {base..base+7} and {base+16..base+23}, base = half*8.
    int bn0   = lane & 15;
    int bbase = half * 8;
    int col   = lane & 15;
    int row0  = wid * 16 + half * 8;

    for (int nT = 0; nT < nTiles; ++nT) {
        const __bf16* bp = Wt + (size_t)(nT * 16 + bn0) * 128 + bbase;
        v8f c = {};
#pragma unroll
        for (int kk = 0; kk < 4; ++kk) {
            v8bf blo = *(const v8bf*)(bp + kk * 32);
            v8bf bhi = *(const v8bf*)(bp + kk * 32 + 16);
            v16bf b;
#pragma unroll
            for (int i = 0; i < 8; ++i) { b[i] = blo[i]; b[i + 8] = bhi[i]; }
            c = __builtin_amdgcn_wmma_f32_16x16x32_bf16(false, a[kk], false, b,
                                                        (short)0, c, false, false);
        }
        float* cp = C + (size_t)row0 * colsPad + nT * 16 + col;
#pragma unroll
        for (int r = 0; r < 8; ++r) cp[(size_t)r * colsPad] = c[r];
    }
}

// ---------------- edge scatter: acc[dst] += norm[e] * t[src], 4 feats/thread ----------------

__global__ void scatter_kernel(const int* __restrict__ ei, const float* __restrict__ nrm,
                               const float* __restrict__ t, float* __restrict__ acc,
                               int Cq, int C, int total) {
    int i = blockIdx.x * blockDim.x + threadIdx.x;
    if (i >= total) return;
    int e = i / Cq;
    int q = i - e * Cq;
    int s = ei[e];
    int d = ei[EE + e];
    float w = nrm[e];
    v4f tv = *(const v4f*)(t + (size_t)s * C + q * 4);
    float* ap = acc + (size_t)d * C + q * 4;
    atomicAdd(ap + 0, w * tv[0]);
    atomicAdd(ap + 1, w * tv[1]);
    atomicAdd(ap + 2, w * tv[2]);
    atomicAdd(ap + 3, w * tv[3]);
}

// ---------------- epilogue: h = relu(acc + dinv^2 * t + b) -> bf16, 4 feats/thread ----------------

__global__ void epilogue_kernel(const float* __restrict__ acc, const float* __restrict__ t,
                                const float* __restrict__ dinv, const float* __restrict__ b,
                                __bf16* __restrict__ h) {
    int i = blockIdx.x * blockDim.x + threadIdx.x;  // quad index
    if (i >= NN * NH / 4) return;
    int node = i >> 5;          // 32 quads per node (NH=128)
    int q    = i & 31;
    float di = dinv[node];
    float d2 = di * di;
    v4f av = *(const v4f*)(acc + (size_t)i * 4);
    v4f tv = *(const v4f*)(t + (size_t)i * 4);
    v4f bv = *(const v4f*)(b + q * 4);
    v4bf hv;
#pragma unroll
    for (int j = 0; j < 4; ++j)
        hv[j] = (__bf16)fmaxf(av[j] + d2 * tv[j] + bv[j], 0.0f);
    *(v4bf*)(h + (size_t)i * 4) = hv;
}

// ---------------- final layer: bias + log_softmax (40 classes), one wave per node ----------------

__global__ void final_kernel(const float* __restrict__ acc, const float* __restrict__ t,
                             const float* __restrict__ dinv, const float* __restrict__ b,
                             float* __restrict__ out) {
    int lane = threadIdx.x & 31;
    int node = (blockIdx.x * blockDim.x + threadIdx.x) >> 5;
    if (node >= NN) return;  // wave-uniform
    float di = dinv[node];
    float d2 = di * di;
    size_t base = (size_t)node * NCPAD;
    float z0 = -__builtin_inff(), z1 = -__builtin_inff();
    if (lane < NC)      z0 = acc[base + lane]      + d2 * t[base + lane]      + b[lane];
    if (lane + 32 < NC) z1 = acc[base + 32 + lane] + d2 * t[base + 32 + lane] + b[32 + lane];
    float m = fmaxf(z0, z1);
#pragma unroll
    for (int off = 16; off >= 1; off >>= 1) m = fmaxf(m, __shfl_xor(m, off, 32));
    float s = (lane < NC ? expf(z0 - m) : 0.0f) + (lane + 32 < NC ? expf(z1 - m) : 0.0f);
#pragma unroll
    for (int off = 16; off >= 1; off >>= 1) s += __shfl_xor(s, off, 32);
    float ls = logf(s) + m;
    if (lane < NC)      out[(size_t)node * NC + lane]      = z0 - ls;
    if (lane + 32 < NC) out[(size_t)node * NC + 32 + lane] = z1 - ls;
}

// ---------------- launch ----------------

extern "C" void kernel_launch(void* const* d_in, const int* in_sizes, int n_in,
                              void* d_out, int out_size, void* d_ws, size_t ws_size,
                              hipStream_t stream) {
    const float* x  = (const float*)d_in[0];
    const int*   ei = (const int*)d_in[1];
    const float* W1 = (const float*)d_in[2];
    const float* b1 = (const float*)d_in[3];
    const float* W2 = (const float*)d_in[4];
    const float* b2 = (const float*)d_in[5];
    const float* W3 = (const float*)d_in[6];
    const float* b3 = (const float*)d_in[7];
    float* out = (float*)d_out;

    // workspace carve
    char* w = (char*)d_ws;
    float*  dinv = (float*)w;  w += alignup((size_t)NN * 4);
    float*  nrm  = (float*)w;  w += alignup((size_t)EE * 4);
    __bf16* hbf  = (__bf16*)w; w += alignup((size_t)NN * NH * 2);
    float*  t    = (float*)w;  w += alignup((size_t)NN * NH * 4);
    float*  acc  = (float*)w;  w += alignup((size_t)NN * NH * 4);
    __bf16* wt1  = (__bf16*)w; w += alignup((size_t)128 * 128 * 2);
    __bf16* wt2  = (__bf16*)w; w += alignup((size_t)128 * 128 * 2);
    __bf16* wt3  = (__bf16*)w; w += alignup((size_t)NCPAD * 128 * 2);

    const int B = 256;

    // normalization: deg (self-loop -> init 1), dinv in place, per-edge norm
    fill_f32<<<cdiv(NN, B), B, 0, stream>>>(dinv, 1.0f, NN);
    deg_kernel<<<cdiv(EE, B), B, 0, stream>>>(ei, dinv);
    rsqrt_kernel<<<cdiv(NN, B), B, 0, stream>>>(dinv, NN);
    norm_kernel<<<cdiv(EE, B), B, 0, stream>>>(ei, dinv, nrm);

    // operand prep
    cvt_bf16<<<cdiv((long long)NN * NF, B), B, 0, stream>>>(x, hbf, NN * NF);
    cvt_w<<<cdiv(128 * 128, B), B, 0, stream>>>(W1, wt1, NH, NH);
    cvt_w<<<cdiv(128 * 128, B), B, 0, stream>>>(W2, wt2, NH, NH);
    cvt_w<<<cdiv(NCPAD * 128, B), B, 0, stream>>>(W3, wt3, NC, NCPAD);

    const int mT = NN / 16;  // 3125 strip-waves per GEMM

    // ----- layer 1 -----
    gemm_wmma<<<cdiv((long long)mT * 32, 128), 128, 0, stream>>>(hbf, wt1, t, mT, NH / 16, NH);
    fill_f32<<<cdiv((long long)NN * NH, B), B, 0, stream>>>(acc, 0.0f, NN * NH);
    scatter_kernel<<<cdiv((long long)EE * (NH / 4), B), B, 0, stream>>>(ei, nrm, t, acc, NH / 4, NH, EE * (NH / 4));
    epilogue_kernel<<<cdiv((long long)NN * NH / 4, B), B, 0, stream>>>(acc, t, dinv, b1, hbf);

    // ----- layer 2 -----
    gemm_wmma<<<cdiv((long long)mT * 32, 128), 128, 0, stream>>>(hbf, wt2, t, mT, NH / 16, NH);
    fill_f32<<<cdiv((long long)NN * NH, B), B, 0, stream>>>(acc, 0.0f, NN * NH);
    scatter_kernel<<<cdiv((long long)EE * (NH / 4), B), B, 0, stream>>>(ei, nrm, t, acc, NH / 4, NH, EE * (NH / 4));
    epilogue_kernel<<<cdiv((long long)NN * NH / 4, B), B, 0, stream>>>(acc, t, dinv, b2, hbf);

    // ----- layer 3 + log_softmax -----
    gemm_wmma<<<cdiv((long long)mT * 32, 128), 128, 0, stream>>>(hbf, wt3, t, mT, NCPAD / 16, NCPAD);
    fill_f32<<<cdiv((long long)NN * NCPAD, B), B, 0, stream>>>(acc, 0.0f, NN * NCPAD);
    scatter_kernel<<<cdiv((long long)EE * (NCPAD / 4), B), B, 0, stream>>>(ei, nrm, t, acc, NCPAD / 4, NCPAD, EE * (NCPAD / 4));
    final_kernel<<<cdiv((long long)NN * 32, 128), 128, 0, stream>>>(acc, t, dinv, b3, out);
}
